// UDLRatingCTM_2576980377844
// MI455X (gfx1250) — matrix operand order repaired and checked by reading, not compiled, not run on hardware
//
#include <hip/hip_runtime.h>
#include <hip/hip_bf16.h>
#include <math.h>

// ---------------------------------------------------------------------------
// UDLRatingCTM forward for MI455X (gfx1250, wave32).
//
// Roofline: ~35 GFLOP total, hot set (Wcomb 21MB bf16 + W1t 10MB + K/V 34MB
// bf16 + emb 65MB) < 192MB L2 -> L2-resident, latency bound. Strategy:
//   * bf16 v_wmma_f32_16x16x32_bf16 for KV projection and the fused synapse
//     GEMM (M=32 = two 16x16 accumulator tiles / wave, fp32 accumulate).
//   * One-time weight fusions: Wcomb = [Wsyn[:, :512] @ Wattn_o | Wsyn[:,512:]]
//     and Wqf = Wpq @ Wq, removing 2 GEMMs from every sequential tick.
//   * One-time bf16 conversion + W1/W2 transposes so per-tick loads stream
//     contiguously from L2.
//   * Attention reads K/V via 128-bit vector loads (v8bf) - K/V traversal is
//     the largest per-tick traffic (~34MB/tick from L2).
// ---------------------------------------------------------------------------

#define B 32
#define S 512
#define D_IN 512
#define D_MODEL 2048
#define HEADS 8
#define HD 64
#define N_OUT 64
#define N_ACT 32
#define M_TR 10
#define MH 64
#define T_TICKS 20
#define K_SYN (D_IN + D_MODEL)  /* 2560 */
#define N_SYN (2 * D_MODEL)     /* 4096 */

typedef __attribute__((ext_vector_type(16))) __bf16 v16bf;
typedef __attribute__((ext_vector_type(8)))  __bf16 v8bf;
typedef __attribute__((ext_vector_type(8)))  float  v8f;

static __device__ __forceinline__ __bf16 f2bf(float f) {
  unsigned u = __builtin_bit_cast(unsigned, f);
  unsigned r = u + 0x7FFFu + ((u >> 16) & 1u);  // round-to-nearest-even
  unsigned short h = (unsigned short)(r >> 16);
  return __builtin_bit_cast(__bf16, h);
}
static __device__ __forceinline__ float bf2f(__bf16 b) {
  unsigned short h = __builtin_bit_cast(unsigned short, b);
  unsigned u = ((unsigned)h) << 16;
  return __builtin_bit_cast(float, u);
}
static __device__ __forceinline__ float sigmoidf_(float x) {
  return 1.0f / (1.0f + expf(-x));
}

union ABf { v16bf v; unsigned u[8]; };

// ---------------------------------------------------------------------------
// Wave-level GEMM: Out[0:32, col:col+16] += A[0:32, 0:K] * W[col][0:K]^T
// A: 32 x K bf16 row-major (lda).  W: N x K bf16 row-major (weight[n][k]).
// Two 16x16xK accumulations per wave (rows 0-15 and 16-31) sharing one B tile.
// VGPR striping per CDNA5 ISA 7.12.2:
//   A 16x32 bf16: lane half h=lane>>4, row r=lane&15; K(j)=j+8h+(j>=8?8:0)
//   B 32x16 bf16: col=lane&15; K(j)=j+8h+(j>=8?8:0) (same as A)
//   C/D f32:      VGPR rr -> M=rr+8h, N=lane&15
// ---------------------------------------------------------------------------
template <bool OUT_BF16>
static __device__ __forceinline__ void wave_gemm_32xKx16(
    const __bf16* __restrict__ A, int lda,
    const __bf16* __restrict__ W, int ldw,
    const float* __restrict__ bias,
    void* __restrict__ Out, int ldo,
    int K, int colbase) {
  const int lane = threadIdx.x & 31;
  const int half = lane >> 4;
  const int r = lane & 15;
  v8f c0 = {};
  v8f c1 = {};
  for (int kk = 0; kk < K; kk += 32) {
    ABf a0, a1, bw;
    const __bf16* Ar0 = A + (size_t)r * lda + kk;
    const __bf16* Ar1 = Ar0 + (size_t)16 * lda;
#pragma unroll
    for (int p = 0; p < 8; ++p) {
      int koff = 2 * p + ((p >= 4) ? 8 : 0) + 8 * half;  // bf16 pair offset
      a0.u[p] = *(const unsigned*)(Ar0 + koff);
      a1.u[p] = *(const unsigned*)(Ar1 + koff);
    }
    const __bf16* Wr = W + (size_t)(colbase + r) * ldw + kk + 8 * half;
#pragma unroll
    for (int p = 0; p < 8; ++p) bw.u[p] = *(const unsigned*)(Wr + 2 * p + ((p >= 4) ? 8 : 0));
    c0 = __builtin_amdgcn_wmma_f32_16x16x32_bf16(false, a0.v, false, bw.v,
                                                 (short)0, c0, false, false);
    c1 = __builtin_amdgcn_wmma_f32_16x16x32_bf16(false, a1.v, false, bw.v,
                                                 (short)0, c1, false, false);
  }
  const int col = colbase + r;
  const float bv = bias ? bias[col] : 0.0f;
#pragma unroll
  for (int rr = 0; rr < 8; ++rr) {
    int row0 = rr + 8 * half;
    int row1 = 16 + row0;
    float v0 = c0[rr] + bv;
    float v1 = c1[rr] + bv;
    if constexpr (OUT_BF16) {
      ((__bf16*)Out)[(size_t)row0 * ldo + col] = f2bf(v0);
      ((__bf16*)Out)[(size_t)row1 * ldo + col] = f2bf(v1);
    } else {
      ((float*)Out)[(size_t)row0 * ldo + col] = v0;
      ((float*)Out)[(size_t)row1 * ldo + col] = v1;
    }
  }
}

// ------------------------- one-time preprocessing --------------------------

__global__ void k_embed(const int* __restrict__ token_ids,
                        const float* __restrict__ emb, __bf16* __restrict__ x) {
  size_t idx = (size_t)blockIdx.x * 256 + threadIdx.x;
  if (idx >= (size_t)B * S * D_IN) return;
  size_t i = idx >> 9;
  int k = (int)(idx & 511);
  int tok = token_ids[i];
  x[idx] = f2bf(emb[(size_t)tok * D_IN + k]);
}

__global__ void k_conv_wkv(const float* __restrict__ Win, __bf16* __restrict__ Wk,
                           __bf16* __restrict__ Wv) {
  size_t idx = (size_t)blockIdx.x * 256 + threadIdx.x;
  if (idx >= (size_t)2 * D_IN * D_IN) return;
  int sel = (int)(idx >> 18);                       // 0=k (rows 512..), 1=v
  size_t rem = idx & (((size_t)1 << 18) - 1);
  __bf16 v = f2bf(Win[(size_t)D_IN * D_IN * (1 + sel) + rem]);
  (sel ? Wv : Wk)[rem] = v;
}

// Wqf = Wpq @ Wq  (512 x 32);  bqf = bpq + Wpq @ bq
__global__ void k_qfuse(const float* __restrict__ Win, const float* __restrict__ b_in,
                        const float* __restrict__ Wq, const float* __restrict__ bq,
                        float* __restrict__ Wqf, float* __restrict__ bqf) {
  const int n = blockIdx.x, c = threadIdx.x;  // 512 blocks x 32 threads
  float acc = 0.f;
  for (int j = 0; j < D_IN; ++j)
    acc += Win[(size_t)n * D_IN + j] * Wq[(size_t)j * N_ACT + c];
  Wqf[n * N_ACT + c] = acc;
  if (c == 0) {
    float bb = b_in[n];
    for (int j = 0; j < D_IN; ++j) bb += Win[(size_t)n * D_IN + j] * bq[j];
    bqf[n] = bb;
  }
}

// Wcomb[n][0:512] = Wsyn[n][0:512] @ Wattn_o ; Wcomb[n][512:] = Wsyn[n][512:]
// bfuse[n] = bsyn[n] + Wsyn[n][0:512] . battn_o
__global__ void k_wcomb(const float* __restrict__ Wsyn, const float* __restrict__ bsyn,
                        const float* __restrict__ Wattn_o,
                        const float* __restrict__ battn_o,
                        __bf16* __restrict__ Wcomb, float* __restrict__ bfuse) {
  __shared__ float wl[D_IN];
  const int n = blockIdx.x, t = threadIdx.x;  // 4096 blocks x 128 threads
  for (int j = t; j < D_IN; j += 128) wl[j] = Wsyn[(size_t)n * K_SYN + j];
  __syncthreads();
  for (int j = t; j < D_IN; j += 128) {
    float acc = 0.f;
    for (int k = 0; k < D_IN; ++k) acc += wl[k] * Wattn_o[(size_t)k * D_IN + j];
    Wcomb[(size_t)n * K_SYN + j] = f2bf(acc);
  }
  for (int k = t; k < D_MODEL; k += 128)
    Wcomb[(size_t)n * K_SYN + D_IN + k] = f2bf(Wsyn[(size_t)n * K_SYN + D_IN + k]);
  if (t == 0) {
    float acc = bsyn[n];
    for (int k = 0; k < D_IN; ++k) acc += wl[k] * battn_o[k];
    bfuse[n] = acc;
  }
}

__global__ void k_w1t(const float* __restrict__ W1, float* __restrict__ W1t) {
  size_t idx = (size_t)blockIdx.x * 256 + threadIdx.x;
  if (idx >= (size_t)M_TR * 2 * MH * D_MODEL) return;
  int n = (int)(idx / (M_TR * 2 * MH));
  int rem = (int)(idx % (M_TR * 2 * MH));
  int m = rem / (2 * MH), j = rem % (2 * MH);
  W1t[idx] = W1[((size_t)m * (2 * MH) + j) * D_MODEL + n];
}

__global__ void k_w2t(const float* __restrict__ W2, float* __restrict__ W2t) {
  size_t idx = (size_t)blockIdx.x * 256 + threadIdx.x;
  if (idx >= (size_t)MH * 2 * D_MODEL) return;
  int n = (int)(idx / (2 * MH));
  int rem = (int)(idx % (2 * MH));
  int h = rem >> 1, o = rem & 1;
  W2t[idx] = W2[((size_t)h * 2 + o) * D_MODEL + n];
}

__global__ void k_init(const float* __restrict__ start_trace,
                       const float* __restrict__ start_act,
                       const float* __restrict__ decay_a,
                       const float* __restrict__ decay_o,
                       const int* __restrict__ lo, const int* __restrict__ ro,
                       float* __restrict__ trace, float* __restrict__ act,
                       float* __restrict__ aa, float* __restrict__ ba,
                       float* __restrict__ ao, float* __restrict__ bo,
                       float* __restrict__ r_a, float* __restrict__ r_o) {
  const size_t idx = (size_t)blockIdx.x * 256 + threadIdx.x;
  if (idx < (size_t)B * D_MODEL * M_TR)
    trace[idx] = start_trace[idx % (D_MODEL * M_TR)];
  if (idx < (size_t)B * D_MODEL) act[idx] = start_act[idx & (D_MODEL - 1)];
  if (idx < B * N_ACT) { aa[idx] = 0.f; ba[idx] = 0.f; }
  if (idx < B * N_OUT) {
    int c = (int)(idx & 63);
    ao[idx] = start_act[lo[c]] * start_act[ro[c]];  // alpha from start state
    bo[idx] = 1.0f;
  }
  if (idx < N_ACT) r_a[idx] = expf(-fminf(fmaxf(decay_a[idx], 0.f), 15.f));
  if (idx < N_OUT) r_o[idx] = expf(-fminf(fmaxf(decay_o[idx], 0.f), 15.f));
}

// kh/vh = x @ Wpk.T / x @ Wpv.T   (bf16 in/out, f32 accumulate via WMMA)
__global__ void k_gemm_kv(const __bf16* __restrict__ x, const __bf16* __restrict__ Wk,
                          const __bf16* __restrict__ Wv, const float* __restrict__ b_in,
                          __bf16* __restrict__ kh, __bf16* __restrict__ vh) {
  const int wave = threadIdx.x >> 5;
  const int rowblk = blockIdx.x;
  const int colbase = blockIdx.y * 64 + wave * 16;
  const bool isV = (blockIdx.z != 0);
  wave_gemm_32xKx16<true>(x + (size_t)rowblk * 32 * D_IN, D_IN,
                          isV ? Wv : Wk, D_IN,
                          b_in + (isV ? 2 * D_IN : D_IN),
                          (void*)((isV ? vh : kh) + (size_t)rowblk * 32 * D_IN),
                          D_IN, D_IN, colbase);
}

// ------------------------------ per-tick ops -------------------------------

// sync_a recurrence + fused q projection: qh = sync_a @ Wqf.T + bqf
__global__ void k_sync_q(const float* __restrict__ act, const int* __restrict__ la,
                         const int* __restrict__ ra, const float* __restrict__ r_a,
                         float* __restrict__ aa, float* __restrict__ ba,
                         const float* __restrict__ Wqf, const float* __restrict__ bqf,
                         float* __restrict__ qh) {
  __shared__ float syncA[B * N_ACT];
  const int t = threadIdx.x;
  for (int i = t; i < B * N_ACT; i += 256) {
    int bb = i >> 5, c = i & 31;
    float pa = act[bb * D_MODEL + la[c]] * act[bb * D_MODEL + ra[c]];
    float rr = r_a[c];
    float na = rr * aa[i] + pa;
    float nb = rr * ba[i] + 1.0f;
    aa[i] = na; ba[i] = nb;
    syncA[i] = na * rsqrtf(nb);
  }
  __syncthreads();
  for (int i = t; i < B * D_IN; i += 256) {
    int bb = i >> 9, nn = i & 511;
    float acc = bqf[nn];
#pragma unroll
    for (int c = 0; c < N_ACT; ++c) acc += syncA[bb * N_ACT + c] * Wqf[nn * N_ACT + c];
    qh[i] = acc;
  }
}

// one block per (b,h): scores -> softmax -> weighted sum of V.
// K/V read via 128-bit vector loads (8 bf16 per global_load_b128).
__global__ void k_attn(const float* __restrict__ qh, const __bf16* __restrict__ kh,
                       const __bf16* __restrict__ vh, float* __restrict__ ctx) {
  __shared__ float qv[HD];
  __shared__ float sc[S];
  __shared__ float red[128];
  __shared__ float redv[128 * 8];
  const int t = threadIdx.x;  // 128 threads
  const int b = blockIdx.x >> 3;
  const int h = blockIdx.x & 7;
  if (t < HD) qv[t] = qh[b * D_IN + h * HD + t];
  __syncthreads();
  // ---- scores ----
  float lmax = -3.4e38f;
  for (int s = t; s < S; s += 128) {
    const v8bf* kr = (const v8bf*)(kh + ((size_t)(b * S + s)) * D_IN + h * HD);
    float acc = 0.f;
#pragma unroll
    for (int g = 0; g < 8; ++g) {
      v8bf kv8 = kr[g];
#pragma unroll
      for (int j = 0; j < 8; ++j) acc += qv[g * 8 + j] * bf2f(kv8[j]);
    }
    acc *= 0.125f;  // 1/sqrt(HD)
    sc[s] = acc;
    lmax = fmaxf(lmax, acc);
  }
  red[t] = lmax; __syncthreads();
  for (int off = 64; off > 0; off >>= 1) {
    if (t < off) red[t] = fmaxf(red[t], red[t + off]);
    __syncthreads();
  }
  const float mx = red[0]; __syncthreads();
  float lsum = 0.f;
  for (int s = t; s < S; s += 128) { float e = expf(sc[s] - mx); sc[s] = e; lsum += e; }
  red[t] = lsum; __syncthreads();
  for (int off = 64; off > 0; off >>= 1) {
    if (t < off) red[t] += red[t + off];
    __syncthreads();
  }
  const float inv = 1.0f / red[0]; __syncthreads();
  // ---- ctx: thread owns 8 contiguous d (one v8bf), strides over s ----
  const int dgrp = t & 7;   // d = dgrp*8 .. dgrp*8+7
  const int spart = t >> 3; // 16 s-partitions
  float acc[8] = {0.f, 0.f, 0.f, 0.f, 0.f, 0.f, 0.f, 0.f};
  for (int s = spart; s < S; s += 16) {
    float p = sc[s];
    v8bf vv = *(const v8bf*)(vh + ((size_t)(b * S + s)) * D_IN + h * HD + dgrp * 8);
#pragma unroll
    for (int j = 0; j < 8; ++j) acc[j] += p * bf2f(vv[j]);
  }
#pragma unroll
  for (int j = 0; j < 8; ++j) redv[t * 8 + j] = acc[j];
  __syncthreads();
  if (t < HD) {
    int dg = t >> 3, j = t & 7;
    float ssum = 0.f;
    for (int p = 0; p < 16; ++p) ssum += redv[((p << 3) | dg) * 8 + j];
    ctx[b * D_IN + h * HD + t] = ssum * inv;
  }
}

// pre = bf16(concat([ctx, act]))
__global__ void k_stage_pre(const float* __restrict__ ctx, const float* __restrict__ act,
                            __bf16* __restrict__ pre) {
  int idx = blockIdx.x * 256 + threadIdx.x;
  if (idx >= B * K_SYN) return;
  int b = idx / K_SYN, k = idx % K_SYN;
  float v = (k < D_IN) ? ctx[b * D_IN + k] : act[b * D_MODEL + (k - D_IN)];
  pre[idx] = f2bf(v);
}

// state_pre = pre @ Wcomb.T + bfuse     (32 x 4096, K = 2560, WMMA bf16)
__global__ void k_gemm_syn(const __bf16* __restrict__ pre, const __bf16* __restrict__ Wcomb,
                           const float* __restrict__ bfuse, float* __restrict__ state_pre) {
  const int wave = threadIdx.x >> 5;
  const int colbase = blockIdx.x * 64 + wave * 16;
  wave_gemm_32xKx16<false>(pre, K_SYN, Wcomb, K_SYN, bfuse, (void*)state_pre,
                           N_SYN, K_SYN, colbase);
}

// GLU -> LayerNorm -> shift into trace
__global__ void k_glu_ln_trace(const float* __restrict__ sp, const float* __restrict__ ln_g,
                               const float* __restrict__ ln_b, float* __restrict__ trace) {
  __shared__ float sbuf[D_MODEL];
  __shared__ float red[256];
  __shared__ float red2[256];
  const int t = threadIdx.x, b = blockIdx.x;
  float lsum = 0.f, lsq = 0.f;
  for (int n = t; n < D_MODEL; n += 256) {
    float a = sp[(size_t)b * N_SYN + n];
    float g = sp[(size_t)b * N_SYN + D_MODEL + n];
    float v = a * sigmoidf_(g);
    sbuf[n] = v; lsum += v; lsq += v * v;
  }
  red[t] = lsum; red2[t] = lsq; __syncthreads();
  for (int off = 128; off > 0; off >>= 1) {
    if (t < off) { red[t] += red[t + off]; red2[t] += red2[t + off]; }
    __syncthreads();
  }
  const float mean = red[0] * (1.f / D_MODEL);
  const float var = red2[0] * (1.f / D_MODEL) - mean * mean;
  const float rstd = rsqrtf(var + 1e-5f);
  for (int n = t; n < D_MODEL; n += 256) {
    float v = (sbuf[n] - mean) * rstd * ln_g[n] + ln_b[n];
    float* tr = trace + ((size_t)(b * D_MODEL + n)) * M_TR;
#pragma unroll
    for (int m = 0; m < M_TR - 1; ++m) tr[m] = tr[m + 1];
    tr[M_TR - 1] = v;
  }
}

// per-neuron NLM: (B x 10)@(10 x 128) -> GLU -> (B x 64)@(64 x 2) -> GLU
__global__ void k_nlm(const float* __restrict__ trace, const float* __restrict__ W1t,
                      const float* __restrict__ b1, const float* __restrict__ W2t,
                      const float* __restrict__ b2, float* __restrict__ act) {
  __shared__ float tr[B * M_TR];       // 320
  __shared__ float w1c[M_TR * 2 * MH]; // 1280
  __shared__ float zb[B * 2 * MH];     // 4096
  __shared__ float h1[B * MH];         // 2048
  const int t = threadIdx.x;           // 128 threads
  const int n = blockIdx.x;
  for (int i = t; i < B * M_TR; i += 128) {
    int bb = i / M_TR, m = i % M_TR;
    tr[i] = trace[((size_t)(bb * D_MODEL + n)) * M_TR + m];
  }
  for (int i = t; i < M_TR * 2 * MH; i += 128)
    w1c[i] = W1t[(size_t)n * (M_TR * 2 * MH) + i];
  __syncthreads();
  for (int i = t; i < B * 2 * MH; i += 128) {
    int bb = i >> 7, j = i & 127;
    float acc = b1[n * (2 * MH) + j];
#pragma unroll
    for (int m = 0; m < M_TR; ++m) acc += tr[bb * M_TR + m] * w1c[m * (2 * MH) + j];
    zb[i] = acc;
  }
  __syncthreads();
  for (int i = t; i < B * MH; i += 128) {
    int bb = i >> 6, hh = i & 63;
    h1[i] = zb[bb * 128 + hh] * sigmoidf_(zb[bb * 128 + 64 + hh]);
  }
  __syncthreads();
  if (t < B) {
    float a0 = b2[n * 2 + 0], a1 = b2[n * 2 + 1];
    for (int hh = 0; hh < MH; ++hh) {
      float hv = h1[t * MH + hh];
      a0 += hv * W2t[n * (2 * MH) + hh * 2 + 0];
      a1 += hv * W2t[n * (2 * MH) + hh * 2 + 1];
    }
    act[t * D_MODEL + n] = a0 * sigmoidf_(a1);
  }
}

__global__ void k_out_update(const float* __restrict__ act, const int* __restrict__ lo,
                             const int* __restrict__ ro, const float* __restrict__ r_o,
                             float* __restrict__ ao, float* __restrict__ bo) {
  for (int i = threadIdx.x; i < B * N_OUT; i += 256) {
    int bb = i >> 6, c = i & 63;
    float po = act[bb * D_MODEL + lo[c]] * act[bb * D_MODEL + ro[c]];
    float rr = r_o[c];
    ao[i] = rr * ao[i] + po;
    bo[i] = rr * bo[i] + 1.0f;
  }
}

// out = [ratings(32) | cert(32x2) | sync_o(32x64)]
__global__ void k_final(const float* __restrict__ ao, const float* __restrict__ bo,
                        const float* __restrict__ Wout, const float* __restrict__ bout,
                        float* __restrict__ out) {
  __shared__ float so[B * N_OUT];
  const int t = threadIdx.x;
  for (int i = t; i < B * N_OUT; i += 256) {
    float v = ao[i] * rsqrtf(bo[i]);
    so[i] = v;
    out[B + B * 2 + i] = v;
  }
  __syncthreads();
  if (t < B) {
    float p = bout[0];
    for (int c = 0; c < N_OUT; ++c) p += so[t * N_OUT + c] * Wout[c];
    out[t] = sigmoidf_(p);
    out[B + t * 2 + 0] = 0.0f;  // out_dims==1 -> entropy 0 (safe-divide branch)
    out[B + t * 2 + 1] = 1.0f;
  }
}

// ------------------------------- launcher ----------------------------------

extern "C" void kernel_launch(void* const* d_in, const int* in_sizes, int n_in,
                              void* d_out, int out_size, void* d_ws, size_t ws_size,
                              hipStream_t stream) {
  (void)in_sizes; (void)n_in; (void)out_size; (void)ws_size;
  const int*   token_ids   = (const int*)  d_in[0];
  const float* emb         = (const float*)d_in[1];
  const float* start_trace = (const float*)d_in[2];
  const float* start_act   = (const float*)d_in[3];
  const float* decay_a     = (const float*)d_in[4];
  const float* decay_o     = (const float*)d_in[5];
  const int*   idx_a_l     = (const int*)  d_in[6];
  const int*   idx_a_r     = (const int*)  d_in[7];
  const int*   idx_o_l     = (const int*)  d_in[8];
  const int*   idx_o_r     = (const int*)  d_in[9];
  const float* Wq          = (const float*)d_in[10];
  const float* bq          = (const float*)d_in[11];
  const float* Win         = (const float*)d_in[12];
  const float* b_in        = (const float*)d_in[13];
  const float* Wattn_o     = (const float*)d_in[14];
  const float* battn_o     = (const float*)d_in[15];
  const float* Wsyn        = (const float*)d_in[16];
  const float* bsyn        = (const float*)d_in[17];
  const float* ln_g        = (const float*)d_in[18];
  const float* ln_b        = (const float*)d_in[19];
  const float* W1          = (const float*)d_in[20];
  const float* b1          = (const float*)d_in[21];
  const float* W2          = (const float*)d_in[22];
  const float* b2          = (const float*)d_in[23];
  const float* Wout        = (const float*)d_in[24];
  const float* bout        = (const float*)d_in[25];

  char* w = (char*)d_ws;
  auto carve = [&](size_t bytes) -> char* {
    char* p = w;
    w += (bytes + 255) & ~(size_t)255;
    return p;
  };
  __bf16* x_bf    = (__bf16*)carve((size_t)B * S * D_IN * 2);
  __bf16* kh      = (__bf16*)carve((size_t)B * S * D_IN * 2);
  __bf16* vh      = (__bf16*)carve((size_t)B * S * D_IN * 2);
  __bf16* Wk_bf   = (__bf16*)carve((size_t)D_IN * D_IN * 2);
  __bf16* Wv_bf   = (__bf16*)carve((size_t)D_IN * D_IN * 2);
  __bf16* Wcomb   = (__bf16*)carve((size_t)N_SYN * K_SYN * 2);
  float*  bfuse   = (float*) carve((size_t)N_SYN * 4);
  float*  Wqf     = (float*) carve((size_t)D_IN * N_ACT * 4);
  float*  bqf     = (float*) carve((size_t)D_IN * 4);
  float*  W1t     = (float*) carve((size_t)D_MODEL * M_TR * 2 * MH * 4);
  float*  W2t     = (float*) carve((size_t)D_MODEL * 2 * MH * 4);
  float*  act     = (float*) carve((size_t)B * D_MODEL * 4);
  float*  trace   = (float*) carve((size_t)B * D_MODEL * M_TR * 4);
  float*  aa      = (float*) carve((size_t)B * N_ACT * 4);
  float*  ba      = (float*) carve((size_t)B * N_ACT * 4);
  float*  ao      = (float*) carve((size_t)B * N_OUT * 4);
  float*  bo      = (float*) carve((size_t)B * N_OUT * 4);
  float*  r_a     = (float*) carve((size_t)N_ACT * 4);
  float*  r_o     = (float*) carve((size_t)N_OUT * 4);
  float*  qh      = (float*) carve((size_t)B * D_IN * 4);
  float*  ctx     = (float*) carve((size_t)B * D_IN * 4);
  __bf16* pre     = (__bf16*)carve((size_t)B * K_SYN * 2);
  float*  stpre   = (float*) carve((size_t)B * N_SYN * 4);

  // ---- one-time preprocessing ----
  k_embed<<<(B * S * D_IN + 255) / 256, 256, 0, stream>>>(token_ids, emb, x_bf);
  k_conv_wkv<<<(2 * D_IN * D_IN + 255) / 256, 256, 0, stream>>>(Win, Wk_bf, Wv_bf);
  k_qfuse<<<D_IN, 32, 0, stream>>>(Win, b_in, Wq, bq, Wqf, bqf);
  k_wcomb<<<N_SYN, 128, 0, stream>>>(Wsyn, bsyn, Wattn_o, battn_o, Wcomb, bfuse);
  k_w1t<<<(M_TR * 2 * MH * D_MODEL + 255) / 256, 256, 0, stream>>>(W1, W1t);
  k_w2t<<<(MH * 2 * D_MODEL + 255) / 256, 256, 0, stream>>>(W2, W2t);
  k_init<<<(B * D_MODEL * M_TR + 255) / 256, 256, 0, stream>>>(
      start_trace, start_act, decay_a, decay_o, idx_o_l, idx_o_r,
      trace, act, aa, ba, ao, bo, r_a, r_o);
  k_gemm_kv<<<dim3(B * S / 32, D_IN / 64, 2), 128, 0, stream>>>(
      x_bf, Wk_bf, Wv_bf, b_in, kh, vh);

  // ---- T sequential ticks ----
  for (int t = 0; t < T_TICKS; ++t) {
    k_sync_q<<<1, 256, 0, stream>>>(act, idx_a_l, idx_a_r, r_a, aa, ba, Wqf, bqf, qh);
    k_attn<<<B * HEADS, 128, 0, stream>>>(qh, kh, vh, ctx);
    k_stage_pre<<<(B * K_SYN + 255) / 256, 256, 0, stream>>>(ctx, act, pre);
    k_gemm_syn<<<N_SYN / 64, 128, 0, stream>>>(pre, Wcomb, bfuse, stpre);
    k_glu_ln_trace<<<B, 256, 0, stream>>>(stpre, ln_g, ln_b, trace);
    k_nlm<<<D_MODEL, 128, 0, stream>>>(trace, W1t, b1, W2t, b2, act);
    k_out_update<<<1, 256, 0, stream>>>(act, idx_o_l, idx_o_r, r_o, ao, bo);
  }

  k_final<<<1, 256, 0, stream>>>(ao, bo, Wout, bout, (float*)d_out);
}
